// MultilayerGRU_66168266162918
// MI455X (gfx1250) — matrix operand order repaired and checked by reading, not compiled, not run on hardware
//
#include <hip/hip_runtime.h>
#include <hip/hip_bf16.h>
#include <stdint.h>

// ---------------------------------------------------------------------------
// MultilayerGRU on MI455X (gfx1250), wave32 + WMMA.
// B=32, S=512, IN=256, H=1024, OUT=256, L=2.
//
// Persistent scan kernel: 64 WGs x 128 threads. Each WG owns 16 hidden
// columns; its 4 waves compute the 4 gate GEMM sections for those columns:
//   wave 0: r  = x@Wxr^T + h@Whr^T   (K = D+H)
//   wave 1: z  = x@Wxz^T + h@Whz^T   (K = D+H)
//   wave 2: nx = x@Wxn^T             (K = D)
//   wave 3: nh = h@Whn^T             (K = H)
// Gate accumulators are exchanged through LDS, the GRU update runs inside the
// WG (f32 state persistent in LDS), and only the f16 h broadcast goes through
// L2 (double-buffered) -> exactly ONE grid barrier per timestep.
// Weights (16 cols x 3(D+H) f16 <= 192KB) are LDS-resident for the whole scan.
// ---------------------------------------------------------------------------

typedef _Float16 h16;
typedef __attribute__((ext_vector_type(16))) _Float16 v16h;
typedef __attribute__((ext_vector_type(8)))  float    v8f;

#define B_    32
#define S_    512
#define IN_   256
#define H_    1024
#define OUT_  256
#define NWG   64
#define NTHR  128

union V16U { v16h v; uint4 q[2]; };

// A-fragment loader, 16-bit A 16x32 lane layout. Pointer already includes the
// per-lane klo (= hiHalf*16) offset; reads 16 contiguous halves at p.
__device__ __forceinline__ v16h load_a16(const h16* __restrict__ p) {
  V16U u;
  u.q[0] = *(const uint4*)p;
  u.q[1] = *(const uint4*)(p + 8);
  return u.v;
}

__device__ __forceinline__ void grid_barrier(int* bar, int phase) {
  __syncthreads();
  __threadfence();
  if (threadIdx.x == 0) {
    __hip_atomic_fetch_add(bar, 1, __ATOMIC_ACQ_REL, __HIP_MEMORY_SCOPE_AGENT);
    const int target = phase * NWG;
    while (__hip_atomic_load(bar, __ATOMIC_ACQUIRE, __HIP_MEMORY_SCOPE_AGENT) < target)
      __builtin_amdgcn_s_sleep(2);
  }
  __syncthreads();
}

__global__ __launch_bounds__(NTHR, 1)
void gru_scan_kernel(const h16* __restrict__ xin,      // (B,S,Din) f16 layer input
                     const h16* __restrict__ slab_r,   // (1024 cols)x(Din+H) f16
                     const h16* __restrict__ slab_z,   // (1024 cols)x(Din+H)
                     const h16* __restrict__ slab_nx,  // (1024 cols)x(Din)
                     const h16* __restrict__ slab_nh,  // (1024 cols)x(H)
                     const float* __restrict__ bxr, const float* __restrict__ bxz,
                     const float* __restrict__ bxn,
                     float* __restrict__ hfin,         // (B,H) f32 final hidden
                     h16*   __restrict__ hbuf0,        // (B,H) f16 ping (pre-zeroed)
                     h16*   __restrict__ hbuf1,        // (B,H) f16 pong
                     h16*   __restrict__ hseq,         // (B,S,H) f16 output sequence
                     int Din, int* __restrict__ bar)   // bar pre-zeroed
{
  __shared__ __align__(64) h16   wl[16 * 6144];        // <=192KB weight strip
  __shared__ __align__(64) float xch[4 * 512];         // gate exchange (4 x 32 x 16)
  __shared__ __align__(64) float hsl[512];             // persistent f32 h slice

  const int wg    = blockIdx.x;              // column strip [wg*16, wg*16+16)
  const int tid   = threadIdx.x;
  const int lane  = tid & 31;
  const int w     = tid >> 5;                // wave = gate section 0..3
  const int laneN = lane & 15;
  const int hiH   = lane >> 4;
  const int klo   = hiH * 16;
  const int KxH   = Din + H_;
  const int Ksec  = (w < 2) ? KxH : ((w == 2) ? Din : H_);
  const int secOff = (w == 0) ? 0 : (w == 1) ? 16 * KxH
                   : (w == 2) ? 32 * KxH : 32 * KxH + 16 * Din;
  const h16* slab = (w == 0) ? slab_r : (w == 1) ? slab_z
                  : (w == 2) ? slab_nx : slab_nh;

  // Preload: each wave copies its 16-col section strip into LDS (once).
  {
    const uint4* src = (const uint4*)(slab + (size_t)wg * 16 * Ksec);
    uint4* dst = (uint4*)(wl + secOff);
    const int n16 = (16 * Ksec) / 8;
    for (int i = lane; i < n16; i += 32) dst[i] = src[i];
  }
  for (int i = tid; i < 512; i += NTHR) hsl[i] = 0.0f;
  __syncthreads();

  // Per-lane steady-state pointers (x rows advance by Din per step).
  const h16* ax0 = xin + (size_t)laneN * S_ * Din + klo;         // batch laneN
  const h16* ax1 = xin + (size_t)(laneN + 16) * S_ * Din + klo;  // batch laneN+16
  const size_t hoff0 = (size_t)laneN * H_ + klo;
  const size_t hoff1 = (size_t)(laneN + 16) * H_ + klo;
  const h16* bpx = wl + secOff + (size_t)laneN * Ksec + hiH * 16;          // K=0
  const h16* bph = bpx + ((w < 2) ? Din : 0);                              // h K-base

  for (int t = 0; t < S_; ++t) {
    const h16* hin  = (t & 1) ? hbuf1 : hbuf0;
    h16*       hout = (t & 1) ? hbuf0 : hbuf1;

    v8f acc0 = {};  // batches 0..15
    v8f acc1 = {};  // batches 16..31

    if (w != 3) {   // x-part: sections r, z, nx
      const h16* p0 = ax0; const h16* p1 = ax1; const h16* bp = bpx;
      for (int k = 0; k < Din; k += 32) {
        v16h a0 = load_a16(p0 + k);
        v16h a1 = load_a16(p1 + k);
        v16h b  = *(const v16h*)(bp + k);
        acc0 = __builtin_amdgcn_wmma_f32_16x16x32_f16(false, a0, false, b,
                                                      (short)0, acc0, false, false);
        acc1 = __builtin_amdgcn_wmma_f32_16x16x32_f16(false, a1, false, b,
                                                      (short)0, acc1, false, false);
      }
    }
    if (w != 2) {   // h-part: sections r, z, nh
      const h16* p0 = hin + hoff0; const h16* p1 = hin + hoff1; const h16* bp = bph;
      for (int k = 0; k < H_; k += 32) {
        v16h a0 = load_a16(p0 + k);
        v16h a1 = load_a16(p1 + k);
        v16h b  = *(const v16h*)(bp + k);
        acc0 = __builtin_amdgcn_wmma_f32_16x16x32_f16(false, a0, false, b,
                                                      (short)0, acc0, false, false);
        acc1 = __builtin_amdgcn_wmma_f32_16x16x32_f16(false, a1, false, b,
                                                      (short)0, acc1, false, false);
      }
    }

    // D layout: VGPR v -> batch = v + (lane>=16 ? 8 : 0), col = laneN.
    #pragma unroll
    for (int v = 0; v < 8; ++v) {
      const int b0 = v + hiH * 8;
      xch[w * 512 + b0 * 16 + laneN]        = acc0[v];
      xch[w * 512 + (b0 + 16) * 16 + laneN] = acc1[v];
    }
    __syncthreads();

    // GRU update for this WG's 32x16 slice, entirely in LDS/registers.
    {
      const int e0 = tid * 4;
      #pragma unroll
      for (int i = 0; i < 4; ++i) {
        const int e  = e0 + i;          // 0..511
        const int bb = e >> 4;
        const int jg = wg * 16 + (e & 15);
        const float gr  = xch[e]        + bxr[jg];
        const float gz  = xch[512 + e]  + bxz[jg];
        const float gnx = xch[1024 + e] + bxn[jg];
        const float gnh = xch[1536 + e];
        const float r = 1.0f / (1.0f + __expf(-gr));
        const float z = 1.0f / (1.0f + __expf(-gz));
        const float n = tanhf(gnx + r * gnh);
        const float hnew = (1.0f - z) * n + z * hsl[e];
        hsl[e] = hnew;
        const h16 h6 = (h16)hnew;
        hout[(size_t)bb * H_ + jg] = h6;
        hseq[((size_t)bb * S_ + t) * H_ + jg] = h6;
        if (t == S_ - 1) hfin[(size_t)bb * H_ + jg] = hnew;
      }
    }

    ax0 += Din; ax1 += Din;
    grid_barrier(bar, t + 1);
  }
}

// ---- output projection: (B*S x H) @ (H x OUT), f16 WMMA ----
__global__ __launch_bounds__(NTHR, 4)
void out_proj_kernel(const h16* __restrict__ A,   // (B*S, H) f16
                     const h16* __restrict__ W,   // (OUT, H) f16 (row-major = B cols)
                     const float* __restrict__ bo,
                     float* __restrict__ out)     // (B*S, OUT) f32
{
  const int lane  = threadIdx.x & 31;
  const int wave  = threadIdx.x >> 5;
  const int laneN = lane & 15;
  const int hiH   = lane >> 4;
  const int klo   = hiH * 16;
  const int mbase = blockIdx.x * 16;
  const int obase = (blockIdx.y * 4 + wave) * 16;

  v8f acc = {};
  const h16* arow = A + (size_t)(mbase + laneN) * H_ + klo;
  const h16* wrow = W + (size_t)(obase + laneN) * H_ + hiH * 16;
  for (int k0 = 0; k0 < H_; k0 += 32) {
    v16h a = load_a16(arow + k0);
    V16U u;
    u.q[0] = *(const uint4*)(wrow + k0);
    u.q[1] = *(const uint4*)(wrow + k0 + 8);
    acc = __builtin_amdgcn_wmma_f32_16x16x32_f16(false, a, false, u.v,
                                                 (short)0, acc, false, false);
  }
  const float bias = bo[obase + laneN];
  #pragma unroll
  for (int v = 0; v < 8; ++v)
    out[(size_t)(mbase + v + hiH * 8) * OUT_ + (obase + laneN)] = acc[v] + bias;
}

// ---- helpers ----
__global__ void cvt_f32_to_f16_kernel(const float* __restrict__ src,
                                      h16* __restrict__ dst, int n) {
  int i = blockIdx.x * blockDim.x + threadIdx.x;
  if (i < n) dst[i] = (h16)src[i];
}

// col j of the combined gate matrix = [Wx[j,:] , Wh[j,:]], K = Din+H contiguous.
__global__ void cat_slab_kernel(const float* __restrict__ Wx,
                                const float* __restrict__ Wh,
                                h16* __restrict__ dst, int Din) {
  const int K = Din + H_;
  int i = blockIdx.x * blockDim.x + threadIdx.x;
  if (i < H_ * K) {
    const int col = i / K, k = i % K;
    const float v = (k < Din) ? Wx[(size_t)col * Din + k]
                              : Wh[(size_t)col * H_ + (k - Din)];
    dst[i] = (h16)v;
  }
}

__global__ void zero32_kernel(uint32_t* __restrict__ p, int n) {
  int i = blockIdx.x * blockDim.x + threadIdx.x;
  if (i < n) p[i] = 0u;
}

// hidden_state output: (B, L=2, H)
__global__ void final_h_kernel(const float* __restrict__ h0,
                               const float* __restrict__ h1,
                               float* __restrict__ dst) {
  int i = blockIdx.x * blockDim.x + threadIdx.x;
  if (i < B_ * 2 * H_) {
    const int b = i / (2 * H_);
    const int rem = i % (2 * H_);
    const int l = rem / H_;
    const int j = rem % H_;
    dst[i] = (l ? h1 : h0)[(size_t)b * H_ + j];
  }
}

extern "C" void kernel_launch(void* const* d_in, const int* in_sizes, int n_in,
                              void* d_out, int out_size, void* d_ws, size_t ws_size,
                              hipStream_t stream) {
  const float* x    = (const float*)d_in[0];
  const float* Wxr0 = (const float*)d_in[1];
  const float* bxr0 = (const float*)d_in[2];
  const float* Whr0 = (const float*)d_in[3];
  const float* Wxz0 = (const float*)d_in[4];
  const float* bxz0 = (const float*)d_in[5];
  const float* Whz0 = (const float*)d_in[6];
  const float* Wxn0 = (const float*)d_in[7];
  const float* bxn0 = (const float*)d_in[8];
  const float* Whn0 = (const float*)d_in[9];
  const float* Wxr1 = (const float*)d_in[10];
  const float* bxr1 = (const float*)d_in[11];
  const float* Whr1 = (const float*)d_in[12];
  const float* Wxz1 = (const float*)d_in[13];
  const float* bxz1 = (const float*)d_in[14];
  const float* Whz1 = (const float*)d_in[15];
  const float* Wxn1 = (const float*)d_in[16];
  const float* bxn1 = (const float*)d_in[17];
  const float* Whn1 = (const float*)d_in[18];
  const float* Wo   = (const float*)d_in[19];
  const float* bo   = (const float*)d_in[20];
  float* out = (float*)d_out;

  uint8_t* ws = (uint8_t*)d_ws;
  size_t off = 0;
  auto take = [&](size_t bytes) -> void* {
    void* p = ws + off;
    off += (bytes + 255) & ~(size_t)255;
    return p;
  };

  h16* xf16   = (h16*)take((size_t)B_ * S_ * IN_ * 2);    // 8 MB
  h16* hseq0  = (h16*)take((size_t)B_ * S_ * H_ * 2);     // 32 MB
  h16* hseq1  = (h16*)take((size_t)B_ * S_ * H_ * 2);     // 32 MB
  h16* sl0_r  = (h16*)take((size_t)H_ * (IN_ + H_) * 2);
  h16* sl0_z  = (h16*)take((size_t)H_ * (IN_ + H_) * 2);
  h16* sl0_nx = (h16*)take((size_t)H_ * IN_ * 2);
  h16* sl0_nh = (h16*)take((size_t)H_ * H_ * 2);
  h16* sl1_r  = (h16*)take((size_t)H_ * (H_ + H_) * 2);
  h16* sl1_z  = (h16*)take((size_t)H_ * (H_ + H_) * 2);
  h16* sl1_nx = (h16*)take((size_t)H_ * H_ * 2);
  h16* sl1_nh = (h16*)take((size_t)H_ * H_ * 2);
  h16* wo16   = (h16*)take((size_t)OUT_ * H_ * 2);
  float* hstate0 = (float*)take((size_t)B_ * H_ * 4);
  float* hstate1 = (float*)take((size_t)B_ * H_ * 4);
  h16*   hbuf0   = (h16*)take((size_t)B_ * H_ * 2);
  h16*   hbuf1   = (h16*)take((size_t)B_ * H_ * 2);
  int*   bar     = (int*)take(256);

  const int T = 256;
  auto blocks = [](int n, int t) { return (n + t - 1) / t; };

  // f32 -> f16 conversions + weight slab packing
  {
    int n = B_ * S_ * IN_;
    cvt_f32_to_f16_kernel<<<blocks(n, T), T, 0, stream>>>(x, xf16, n);
  }
  cat_slab_kernel<<<blocks(H_ * (IN_ + H_), T), T, 0, stream>>>(Wxr0, Whr0, sl0_r, IN_);
  cat_slab_kernel<<<blocks(H_ * (IN_ + H_), T), T, 0, stream>>>(Wxz0, Whz0, sl0_z, IN_);
  cvt_f32_to_f16_kernel<<<blocks(H_ * IN_, T), T, 0, stream>>>(Wxn0, sl0_nx, H_ * IN_);
  cvt_f32_to_f16_kernel<<<blocks(H_ * H_, T), T, 0, stream>>>(Whn0, sl0_nh, H_ * H_);
  cat_slab_kernel<<<blocks(H_ * (H_ + H_), T), T, 0, stream>>>(Wxr1, Whr1, sl1_r, H_);
  cat_slab_kernel<<<blocks(H_ * (H_ + H_), T), T, 0, stream>>>(Wxz1, Whz1, sl1_z, H_);
  cvt_f32_to_f16_kernel<<<blocks(H_ * H_, T), T, 0, stream>>>(Wxn1, sl1_nx, H_ * H_);
  cvt_f32_to_f16_kernel<<<blocks(H_ * H_, T), T, 0, stream>>>(Whn1, sl1_nh, H_ * H_);
  cvt_f32_to_f16_kernel<<<blocks(OUT_ * H_, T), T, 0, stream>>>(Wo, wo16, OUT_ * H_);

  // ---- layer 0 scan ----
  zero32_kernel<<<blocks(B_ * H_ / 2, T), T, 0, stream>>>((uint32_t*)hbuf0, B_ * H_ / 2);
  zero32_kernel<<<1, 64, 0, stream>>>((uint32_t*)bar, 64);
  gru_scan_kernel<<<NWG, NTHR, 0, stream>>>(xf16, sl0_r, sl0_z, sl0_nx, sl0_nh,
                                            bxr0, bxz0, bxn0,
                                            hstate0, hbuf0, hbuf1, hseq0, IN_, bar);

  // ---- layer 1 scan ----
  zero32_kernel<<<blocks(B_ * H_ / 2, T), T, 0, stream>>>((uint32_t*)hbuf0, B_ * H_ / 2);
  zero32_kernel<<<1, 64, 0, stream>>>((uint32_t*)bar, 64);
  gru_scan_kernel<<<NWG, NTHR, 0, stream>>>(hseq0, sl1_r, sl1_z, sl1_nx, sl1_nh,
                                            bxr1, bxz1, bxn1,
                                            hstate1, hbuf0, hbuf1, hseq1, H_, bar);

  // ---- output projection + hidden_state ----
  out_proj_kernel<<<dim3(B_ * S_ / 16, OUT_ / 64), NTHR, 0, stream>>>(hseq1, wo16, bo, out);
  final_h_kernel<<<blocks(B_ * 2 * H_, T), T, 0, stream>>>(hstate0, hstate1,
                                                           out + (size_t)B_ * S_ * OUT_);
}
